// MixtureOfAttention_75041668596076
// MI455X (gfx1250) — matrix operand order repaired and checked by reading, not compiled, not run on hardware
//
#include <hip/hip_runtime.h>
#include <math.h>
#include <stdint.h>

// ---------------------------------------------------------------------------
// MixtureOfAttention for MI455X (gfx1250, wave32, WMMA bf16 16x16x32,
// async global->LDS staging tracked by ASYNCcnt).
//
// Math (after dead-code elimination + associativity):
//   xn      = rmsnorm(x) * norm_w
//   routing = softmax(xn @ router_w + router_b)
//   WcT[e]  = (qkv_w[e][:,1024:1536] @ proj_w[e])^T        (tiny prep, N x K)
//   out     = concat_e( (xn_e @ Wc[e] + proj_b[e]) * routing[:,e] )
//   y       = x + out @ out_w          (out_w pre-transposed to bf16 N x K)
// ---------------------------------------------------------------------------

#define Bdim 32768
#define Ddim 2048
#define Edim 4
#define dEdim 512

typedef __attribute__((ext_vector_type(16))) __bf16 v16bf;
typedef __attribute__((ext_vector_type(8)))  float  v8f;
typedef int v4i_gcc __attribute__((vector_size(16)));   // matches builtin proto

#define AS1 __attribute__((address_space(1)))
#define AS3 __attribute__((address_space(3)))

union FragV {
    uint4 u[2];
    v16bf v;
};

__device__ __forceinline__ unsigned short f2bf(float f) {
    unsigned int u = __float_as_uint(f);
    u += 0x7FFFu + ((u >> 16) & 1u);   // round-to-nearest-even bf16
    return (unsigned short)(u >> 16);
}

// ---- CDNA5 async global->LDS copy (ASYNCcnt) ------------------------------
__device__ __forceinline__ void async_ld_b128(const void* g, void* l) {
#if __has_builtin(__builtin_amdgcn_global_load_async_to_lds_b128)
    __builtin_amdgcn_global_load_async_to_lds_b128(
        (AS1 v4i_gcc*)(uintptr_t)g,
        (AS3 v4i_gcc*)(unsigned)(uintptr_t)l, 0, 0);
#else
    const unsigned laddr = (unsigned)(uintptr_t)l;           // LDS byte offset
    const unsigned long long gaddr = (unsigned long long)(uintptr_t)g;
    asm volatile("global_load_async_to_lds_b128 %0, %1, off"
                 :: "v"(laddr), "v"(gaddr) : "memory");
#endif
}

template <int N>
__device__ __forceinline__ void wait_asynccnt() {
#if __has_builtin(__builtin_amdgcn_s_wait_asynccnt)
    __builtin_amdgcn_s_wait_asynccnt(N);
#else
    asm volatile("s_wait_asynccnt %0" :: "i"(N) : "memory");
#endif
}

// ---------------------------------------------------------------------------
// Kernel 1: per-row RMSNorm + router logits + softmax.
// ---------------------------------------------------------------------------
__global__ __launch_bounds__(256) void rmsnorm_router_kern(
    const float* __restrict__ x, const float* __restrict__ norm_w,
    const float* __restrict__ router_w, const float* __restrict__ router_b,
    unsigned short* __restrict__ xn, float* __restrict__ routing)
{
    __shared__ float sred[256];
    __shared__ float slog[4 * 256];

    const int row = blockIdx.x;
    const int tid = threadIdx.x;
    const float* xr = x + (size_t)row * Ddim;

    float xl[8];
    float sq = 0.0f;
#pragma unroll
    for (int i = 0; i < 8; ++i) {
        float v = xr[tid + i * 256];
        xl[i] = v;
        sq += v * v;
    }
    sred[tid] = sq;
    __syncthreads();
    for (int s = 128; s > 0; s >>= 1) {
        if (tid < s) sred[tid] += sred[tid + s];
        __syncthreads();
    }
    const float inv = rsqrtf(sred[0] * (1.0f / (float)Ddim) + 1e-6f);

    float l0 = 0.f, l1 = 0.f, l2 = 0.f, l3 = 0.f;
#pragma unroll
    for (int i = 0; i < 8; ++i) {
        const int d = tid + i * 256;
        const float v = xl[i] * inv * norm_w[d];
        xn[(size_t)row * Ddim + d] = f2bf(v);
        const float4 w = ((const float4*)router_w)[d];   // (D,4) row-major
        l0 += v * w.x; l1 += v * w.y; l2 += v * w.z; l3 += v * w.w;
    }
    slog[tid] = l0; slog[256 + tid] = l1; slog[512 + tid] = l2; slog[768 + tid] = l3;
    __syncthreads();
    for (int s = 128; s > 0; s >>= 1) {
        if (tid < s) {
            slog[tid]       += slog[tid + s];
            slog[256 + tid] += slog[256 + tid + s];
            slog[512 + tid] += slog[512 + tid + s];
            slog[768 + tid] += slog[768 + tid + s];
        }
        __syncthreads();
    }
    if (tid == 0) {
        float lg[4];
#pragma unroll
        for (int e = 0; e < 4; ++e) lg[e] = slog[e * 256] + router_b[e];
        float mx = fmaxf(fmaxf(lg[0], lg[1]), fmaxf(lg[2], lg[3]));
        float p[4], s = 0.f;
#pragma unroll
        for (int e = 0; e < 4; ++e) { p[e] = expf(lg[e] - mx); s += p[e]; }
        const float rs = 1.0f / s;
#pragma unroll
        for (int e = 0; e < 4; ++e) routing[(size_t)row * 4 + e] = p[e] * rs;
    }
}

// ---------------------------------------------------------------------------
// Kernel 2: tiled transpose + f32->bf16:  dst[n*N + k] = bf16(src[k*N + n])
// ---------------------------------------------------------------------------
__global__ __launch_bounds__(256) void transpose_bf16_kern(
    const float* __restrict__ src, unsigned short* __restrict__ dst, int n)
{
    __shared__ float t[32][33];
    const int tx = threadIdx.x & 31;
    const int ty = threadIdx.x >> 5;
    const int c0 = blockIdx.x * 32;
    const int r0 = blockIdx.y * 32;
#pragma unroll
    for (int i = 0; i < 4; ++i)
        t[ty + i * 8][tx] = src[(size_t)(r0 + ty + i * 8) * n + c0 + tx];
    __syncthreads();
#pragma unroll
    for (int i = 0; i < 4; ++i)
        dst[(size_t)(c0 + ty + i * 8) * n + r0 + tx] = f2bf(t[tx][ty + i * 8]);
}

// ---------------------------------------------------------------------------
// Kernel 3 (prep): WcT[e] = (Wv[e] @ proj_w[e])^T, f32 in (manual convert),
// bf16 out stored transposed (N x K). Tiny: 4 x 512^3 FLOPs, 64 blocks.
// 256 thr = 8 waves, tile 128x128x32, wave 64x32 (4x2 WMMA accums).
// ---------------------------------------------------------------------------
__global__ __launch_bounds__(256) void gemm_prep_kern(
    const float* __restrict__ qkv_w, const float* __restrict__ proj_w,
    unsigned short* __restrict__ WcT)
{
    constexpr int LDSP = 40;
    __shared__ unsigned short lds_a[128 * LDSP];
    __shared__ unsigned short lds_b[128 * LDSP];

    const int tid  = threadIdx.x;
    const int lane = tid & 31;
    const int wave = tid >> 5;
    const int wm = wave >> 2, wn = wave & 3;
    const int m0 = blockIdx.x * 128;
    const int n0 = blockIdx.y * 128;
    const int e  = blockIdx.z;

    const float* Af = qkv_w + (size_t)e * dEdim * (3 * dEdim) + 2 * dEdim; // V slice
    const float* Bf = proj_w + (size_t)e * dEdim * dEdim;

    v8f acc[4][2];
#pragma unroll
    for (int i = 0; i < 4; ++i)
#pragma unroll
        for (int j = 0; j < 2; ++j)
            acc[i][j] = (v8f){0.f, 0.f, 0.f, 0.f, 0.f, 0.f, 0.f, 0.f};

    const int ar   = tid >> 1;
    const int aseg = (tid & 1) * 16;
    const int bk   = tid & 31;
    const int bn   = (tid >> 5) * 16;

    for (int k0 = 0; k0 < dEdim; k0 += 32) {
        {   // A panel (f32 -> bf16), row-major [m][k]
            const float* src = Af + (size_t)(m0 + ar) * (3 * dEdim) + k0 + aseg;
#pragma unroll
            for (int i = 0; i < 4; ++i) {
                const float4 f = ((const float4*)src)[i];
                lds_a[ar * LDSP + aseg + i * 4 + 0] = f2bf(f.x);
                lds_a[ar * LDSP + aseg + i * 4 + 1] = f2bf(f.y);
                lds_a[ar * LDSP + aseg + i * 4 + 2] = f2bf(f.z);
                lds_a[ar * LDSP + aseg + i * 4 + 3] = f2bf(f.w);
            }
        }
        {   // B panel (f32 -> bf16), transposed to [n][k]
            const float* src = Bf + (size_t)(k0 + bk) * dEdim + n0 + bn;
#pragma unroll
            for (int i = 0; i < 4; ++i) {
                const float4 f = ((const float4*)src)[i];
                lds_b[(bn + i * 4 + 0) * LDSP + bk] = f2bf(f.x);
                lds_b[(bn + i * 4 + 1) * LDSP + bk] = f2bf(f.y);
                lds_b[(bn + i * 4 + 2) * LDSP + bk] = f2bf(f.z);
                lds_b[(bn + i * 4 + 3) * LDSP + bk] = f2bf(f.w);
            }
        }
        __syncthreads();

        FragV afr[4], bfr[2];
        const int lrow = lane & 15;
        const int akb  = (lane >> 4) * 8;
#pragma unroll
        for (int i = 0; i < 4; ++i) {
            const int r = wm * 64 + i * 16 + lrow;
            afr[i].u[0] = *(const uint4*)&lds_a[r * LDSP + akb];
            afr[i].u[1] = *(const uint4*)&lds_a[r * LDSP + akb + 16];
        }
        const int bkb = (lane >> 4) * 8;
#pragma unroll
        for (int j = 0; j < 2; ++j) {
            const int nn = wn * 32 + j * 16 + lrow;
            bfr[j].u[0] = *(const uint4*)&lds_b[nn * LDSP + bkb];
            bfr[j].u[1] = *(const uint4*)&lds_b[nn * LDSP + bkb + 16];
        }
#pragma unroll
        for (int i = 0; i < 4; ++i)
#pragma unroll
            for (int j = 0; j < 2; ++j)
                acc[i][j] = __builtin_amdgcn_wmma_f32_16x16x32_bf16(
                    false, afr[i].v, false, bfr[j].v, (short)0, acc[i][j],
                    false, false);
        __syncthreads();
    }

    // store C^T (WcT is N x K)
    const int mlo = (lane >> 4) * 8;
    const int nlo = lane & 15;
#pragma unroll
    for (int i = 0; i < 4; ++i)
#pragma unroll
        for (int j = 0; j < 2; ++j)
#pragma unroll
            for (int v = 0; v < 8; ++v) {
                const int m = m0 + wm * 64 + i * 16 + mlo + v;
                const int n = n0 + wn * 32 + j * 16 + nlo;
                WcT[(size_t)e * dEdim * dEdim + (size_t)n * dEdim + m] =
                    f2bf(acc[i][j][v]);
            }
}

// ---------------------------------------------------------------------------
// Main GEMM: C = A @ B with A (M x K) bf16 row-major and B supplied
// PRE-TRANSPOSED bf16 (N x K row-major), so both LDS panels are straight
// async copies (global_load_async_to_lds_b128), double-buffered on ASYNCcnt.
//
// MODE 1: A = xn (+e*512 col offset), B = WcT[e], epilogue
//         bf16( (acc + proj_b[e][n]) * routing[row][e] ) -> out
// MODE 2: A = out, B = out_wT, epilogue  x[row][n] + acc -> y (f32)
// ---------------------------------------------------------------------------
template <int MODE>
__global__ __launch_bounds__(256) void gemm_main_kern(
    const unsigned short* __restrict__ Ap, int lda,
    const unsigned short* __restrict__ BpT, int ldb,   // N x K row-major
    void* __restrict__ Cp, int Kdim,
    const float* __restrict__ bias,
    const float* __restrict__ routing,
    const float* __restrict__ xres)
{
    constexpr int LDSP = 40;
    __shared__ unsigned short lds_a[2][128 * LDSP];
    __shared__ unsigned short lds_b[2][128 * LDSP];

    const int tid  = threadIdx.x;
    const int lane = tid & 31;
    const int wave = tid >> 5;
    const int wm = wave >> 2, wn = wave & 3;
    const int m0 = blockIdx.x * 128;
    const int n0 = blockIdx.y * 128;
    const int e  = blockIdx.z;

    const unsigned short* Abase = (MODE == 1) ? Ap + (size_t)e * dEdim : Ap;
    const unsigned short* Bbase = (MODE == 1) ? BpT + (size_t)e * dEdim * dEdim : BpT;

    v8f acc[4][2];
#pragma unroll
    for (int i = 0; i < 4; ++i)
#pragma unroll
        for (int j = 0; j < 2; ++j)
            acc[i][j] = (v8f){0.f, 0.f, 0.f, 0.f, 0.f, 0.f, 0.f, 0.f};

    // panel staging: 128 rows x 32 cols bf16, 16 elems (2 x b128) per thread
    const int ar   = tid >> 1;
    const int aseg = (tid & 1) * 16;

    auto stage = [&](int k0, int buf) {
        const unsigned short* asrc = Abase + (size_t)(m0 + ar) * lda + k0 + aseg;
        async_ld_b128(asrc,     &lds_a[buf][ar * LDSP + aseg]);
        async_ld_b128(asrc + 8, &lds_a[buf][ar * LDSP + aseg + 8]);
        const unsigned short* bsrc = Bbase + (size_t)(n0 + ar) * ldb + k0 + aseg;
        async_ld_b128(bsrc,     &lds_b[buf][ar * LDSP + aseg]);
        async_ld_b128(bsrc + 8, &lds_b[buf][ar * LDSP + aseg + 8]);
    };

    stage(0, 0);
    const int iters = Kdim / 32;
    for (int it = 0; it < iters; ++it) {
        const int cur = it & 1;
        const bool more = (it + 1) < iters;
        if (more) {
            stage((it + 1) * 32, cur ^ 1);
            wait_asynccnt<4>();   // current panel's 4 copies have landed
        } else {
            wait_asynccnt<0>();
        }
        __syncthreads();

        FragV afr[4], bfr[2];
        const int lrow = lane & 15;
        // A operand: lanes 0-15 -> K 0-7 & 16-23; lanes 16-31 -> K 8-15 & 24-31
        const int akb  = (lane >> 4) * 8;
#pragma unroll
        for (int i = 0; i < 4; ++i) {
            const int r = wm * 64 + i * 16 + lrow;
            afr[i].u[0] = *(const uint4*)&lds_a[cur][r * LDSP + akb];
            afr[i].u[1] = *(const uint4*)&lds_a[cur][r * LDSP + akb + 16];
        }
        // B operand: lanes 0-15 -> K 0-7 & 16-23; lanes 16-31 -> K 8-15 & 24-31
        const int bkb = (lane >> 4) * 8;
#pragma unroll
        for (int j = 0; j < 2; ++j) {
            const int nn = wn * 32 + j * 16 + lrow;
            bfr[j].u[0] = *(const uint4*)&lds_b[cur][nn * LDSP + bkb];
            bfr[j].u[1] = *(const uint4*)&lds_b[cur][nn * LDSP + bkb + 16];
        }
#pragma unroll
        for (int i = 0; i < 4; ++i)
#pragma unroll
            for (int j = 0; j < 2; ++j)
                acc[i][j] = __builtin_amdgcn_wmma_f32_16x16x32_bf16(
                    false, afr[i].v, false, bfr[j].v, (short)0, acc[i][j],
                    false, false);
        __syncthreads();
    }

    const int mlo = (lane >> 4) * 8;
    const int nlo = lane & 15;
#pragma unroll
    for (int i = 0; i < 4; ++i)
#pragma unroll
        for (int j = 0; j < 2; ++j)
#pragma unroll
            for (int v = 0; v < 8; ++v) {
                const int m = m0 + wm * 64 + i * 16 + mlo + v;
                const int n = wn * 32 + j * 16 + nlo;
                const float a = acc[i][j][v];
                if constexpr (MODE == 1) {
                    const float r = routing[(size_t)m * 4 + e];
                    const float b = bias[(size_t)e * dEdim + n0 + n];
                    ((unsigned short*)Cp)[(size_t)m * Ddim + (size_t)e * dEdim + n0 + n] =
                        f2bf((a + b) * r);
                } else {
                    const size_t idx = (size_t)m * Ddim + n0 + n;
                    ((float*)Cp)[idx] = xres[idx] + a;
                }
            }
}

// ---------------------------------------------------------------------------
extern "C" void kernel_launch(void* const* d_in, const int* in_sizes, int n_in,
                              void* d_out, int out_size, void* d_ws, size_t ws_size,
                              hipStream_t stream) {
    const float* x        = (const float*)d_in[0];
    const float* norm_w   = (const float*)d_in[1];
    const float* router_w = (const float*)d_in[2];
    const float* router_b = (const float*)d_in[3];
    const float* qkv_w    = (const float*)d_in[4];
    const float* proj_w   = (const float*)d_in[5];
    const float* proj_b   = (const float*)d_in[6];
    const float* out_w    = (const float*)d_in[7];
    float* y = (float*)d_out;

    // workspace layout (256B aligned)
    char* ws = (char*)d_ws;
    unsigned short* xn_bf  = (unsigned short*)(ws);                              // 128MB
    unsigned short* out_bf = (unsigned short*)(ws + (size_t)Bdim * Ddim * 2);    // 128MB
    float* routing = (float*)(ws + (size_t)2 * Bdim * Ddim * 2);                 // 512KB
    unsigned short* WcT = (unsigned short*)((char*)routing + (size_t)Bdim * 16); // 2MB
    unsigned short* out_wT = WcT + (size_t)Edim * dEdim * dEdim;                 // 8MB

    // 1) WcT[e] = (Wv[e] @ proj_w[e])^T   (tiny)
    gemm_prep_kern<<<dim3(dEdim / 128, dEdim / 128, Edim), 256, 0, stream>>>(
        qkv_w, proj_w, WcT);

    // 2) out_wT = transpose(out_w) as bf16
    transpose_bf16_kern<<<dim3(Ddim / 32, Ddim / 32), 256, 0, stream>>>(
        out_w, out_wT, Ddim);

    // 3) rmsnorm + router softmax, xn -> bf16
    rmsnorm_router_kern<<<Bdim, 256, 0, stream>>>(
        x, norm_w, router_w, router_b, xn_bf, routing);

    // 4) out[:, e*512:+512] = (xn_e @ Wc[e] + proj_b[e]) * routing[:, e]
    gemm_main_kern<1><<<dim3(Bdim / 128, dEdim / 128, Edim), 256, 0, stream>>>(
        xn_bf, Ddim, WcT, dEdim, out_bf, dEdim, proj_b, routing, nullptr);

    // 5) y = x + out @ out_w
    gemm_main_kern<2><<<dim3(Bdim / 128, Ddim / 128, 1), 256, 0, stream>>>(
        out_bf, Ddim, out_wT, Ddim, y, Ddim, nullptr, nullptr, x);
}
